// ElmanRNN_78451872629440
// MI455X (gfx1250) — matrix operand order repaired and checked by reading, not compiled, not run on hardware
//
#include <hip/hip_runtime.h>
#include <hip/hip_bf16.h>
#include <math.h>

#define B_ 64
#define S_ 1024
#define I_ 512
#define H_ 512

typedef __attribute__((ext_vector_type(16))) __bf16 v16bf;
typedef __attribute__((ext_vector_type(8)))  float  v8f;
typedef __attribute__((ext_vector_type(4)))  float  f32x4;

__device__ __forceinline__ float fast_tanh(float v) {
#if __has_builtin(__builtin_amdgcn_tanhf)
    return __builtin_amdgcn_tanhf(v);
#else
    return tanhf(v);
#endif
}

// ---------------------------------------------------------------------------
// Kernel 0: one-time prep. Convert W_ih / W_hh to bf16 in workspace, fold the
// two bias vectors into one f32 vector. ~1 MB of workspace total.
// ---------------------------------------------------------------------------
__global__ void elman_prep_kernel(const float* __restrict__ Wih,
                                  const float* __restrict__ Whh,
                                  const float* __restrict__ bih,
                                  const float* __restrict__ bhh,
                                  __bf16* __restrict__ Wih_bf,
                                  __bf16* __restrict__ Whh_bf,
                                  float* __restrict__ bias) {
    int i = blockIdx.x * blockDim.x + threadIdx.x;
    if (i < H_ * I_) {
        Wih_bf[i] = (__bf16)Wih[i];
        Whh_bf[i] = (__bf16)Whh[i];
    }
    if (i < H_) bias[i] = bih[i] + bhh[i];
}

// Build a 16x32 bf16 A-fragment for v_wmma_f32_16x16x32_bf16 from a row-major
// f32 row pointer. Contiguous K order matching B: lanes 0-15 hold K in {0..15},
// lanes 16-31 hold K in {16..31} (offset 2*koffA = 16*(lane>=16)).
__device__ __forceinline__ v16bf make_a_frag(const float* __restrict__ row,
                                             int k0, int koffA) {
    f32x4 a0 = *(const f32x4*)(row + k0 + 2 * koffA);
    f32x4 a1 = *(const f32x4*)(row + k0 + 2 * koffA + 4);
    f32x4 a2 = *(const f32x4*)(row + k0 + 2 * koffA + 8);
    f32x4 a3 = *(const f32x4*)(row + k0 + 2 * koffA + 12);
    v16bf a;
#pragma unroll
    for (int j = 0; j < 4; ++j) {
        a[j]      = (__bf16)a0[j];
        a[4 + j]  = (__bf16)a1[j];
        a[8 + j]  = (__bf16)a2[j];
        a[12 + j] = (__bf16)a3[j];
    }
    return a;
}

// ---------------------------------------------------------------------------
// Kernel 1: x_proj = x @ W_ih^T + bias, written into d_out (B*S, H) row-major.
// M = B*S = 65536, N = 512, K = 512. Each wave: one 16-row strip x 64 columns
// (one A fragment feeds 4 WMMA tiles per K-step). 8 waves per block tile N.
// Software-pipelined: fragments for k+1 are issued before the k WMMAs, so
// L2 latency hides under the matrix ops instead of stalling each iteration.
// ---------------------------------------------------------------------------
__global__ void __launch_bounds__(256)
elman_xproj_kernel(const float* __restrict__ x,
                   const __bf16* __restrict__ Wih_bf,
                   const float* __restrict__ bias,
                   float* __restrict__ out) {
    const int lane  = threadIdx.x & 31;
    const int wave  = threadIdx.x >> 5;
    const int mtile = blockIdx.x;        // 0..4095, 16 rows of M each
    const int n0    = wave * 64;         // 8 waves cover N=512
    const int lm    = lane & 15;
    const int hi    = lane >> 4;         // half-wave select
    const int koffA = hi * 8;
    const int koffB = hi * 16;

    const float* xrow = x + (size_t)(mtile * 16 + lm) * I_;
    // B operand: W_ih is (N=H, K=I) row-major; lane lm owns column n0+16j+lm,
    // needs 16 contiguous K values at offset koffB -> straight 32-byte loads.
    const __bf16* bp = Wih_bf + (size_t)(n0 + lm) * I_ + koffB;

    v8f acc0 = {}, acc1 = {}, acc2 = {}, acc3 = {};

    // prologue: fragments for k0 = 0
    v16bf a  = make_a_frag(xrow, 0, koffA);
    v16bf b0 = *(const v16bf*)(bp);
    v16bf b1 = *(const v16bf*)(bp + 16 * I_);
    v16bf b2 = *(const v16bf*)(bp + 32 * I_);
    v16bf b3 = *(const v16bf*)(bp + 48 * I_);

    for (int k0 = 32; k0 < I_; k0 += 32) {
        // issue next-iteration fetches first
        v16bf an  = make_a_frag(xrow, k0, koffA);
        v16bf bn0 = *(const v16bf*)(bp + k0);
        v16bf bn1 = *(const v16bf*)(bp + 16 * I_ + k0);
        v16bf bn2 = *(const v16bf*)(bp + 32 * I_ + k0);
        v16bf bn3 = *(const v16bf*)(bp + 48 * I_ + k0);
        acc0 = __builtin_amdgcn_wmma_f32_16x16x32_bf16(false, a, false, b0,
                                                       (short)0, acc0, false, false);
        acc1 = __builtin_amdgcn_wmma_f32_16x16x32_bf16(false, a, false, b1,
                                                       (short)0, acc1, false, false);
        acc2 = __builtin_amdgcn_wmma_f32_16x16x32_bf16(false, a, false, b2,
                                                       (short)0, acc2, false, false);
        acc3 = __builtin_amdgcn_wmma_f32_16x16x32_bf16(false, a, false, b3,
                                                       (short)0, acc3, false, false);
        a = an; b0 = bn0; b1 = bn1; b2 = bn2; b3 = bn3;
    }
    acc0 = __builtin_amdgcn_wmma_f32_16x16x32_bf16(false, a, false, b0,
                                                   (short)0, acc0, false, false);
    acc1 = __builtin_amdgcn_wmma_f32_16x16x32_bf16(false, a, false, b1,
                                                   (short)0, acc1, false, false);
    acc2 = __builtin_amdgcn_wmma_f32_16x16x32_bf16(false, a, false, b2,
                                                   (short)0, acc2, false, false);
    acc3 = __builtin_amdgcn_wmma_f32_16x16x32_bf16(false, a, false, b3,
                                                   (short)0, acc3, false, false);

    // C/D layout: vgpr r -> row (r + 8*hi), lane lm -> column.
    const int nc = n0 + lm;
    const float bz0 = bias[nc], bz1 = bias[nc + 16],
                bz2 = bias[nc + 32], bz3 = bias[nc + 48];
#pragma unroll
    for (int r = 0; r < 8; ++r) {
        float* orow = out + (size_t)(mtile * 16 + r + hi * 8) * H_;
        orow[nc]      = acc0[r] + bz0;
        orow[nc + 16] = acc1[r] + bz1;
        orow[nc + 32] = acc2[r] + bz2;
        orow[nc + 48] = acc3[r] + bz3;
    }
}

// ---------------------------------------------------------------------------
// Kernel 2: one recurrence step, in place in d_out (B, S, H):
//   out[:, t, :] = tanh(out[:, t, :] + out[:, t-1, :] @ W_hh^T)
// M = 64, N = 512, K = 512 -> 4 x 32 tiles = 16 blocks x 8 waves.
// All 8 waves of a block share one mtile (same A rows -> WGP$ hits);
// the x_proj term is fetched before the k-loop to overlap with the GEMM.
// t == 0 skips the K-loop uniformly (h_prev = 0), EXEC stays all-ones.
// h (128 KB) and W_hh (512 KB bf16) are L2-resident across all 1024 steps.
// ---------------------------------------------------------------------------
__global__ void __launch_bounds__(256)
elman_step_kernel(const __bf16* __restrict__ Whh_bf,
                  float* __restrict__ out, int t) {
    const int lane  = threadIdx.x & 31;
    const int wave  = threadIdx.x >> 5;
    const int mtile = blockIdx.x & 3;               // 4 tiles cover B=64
    const int ntile = (blockIdx.x >> 2) * 8 + wave; // 32 tiles cover H=512
    const int lm    = lane & 15;
    const int hi    = lane >> 4;
    const int koffA = hi * 8;
    const int koffB = hi * 16;
    const int n     = ntile * 16 + lm;

    // fetch the x_proj contribution up front (overlaps with the GEMM below)
    float xp[8];
#pragma unroll
    for (int r = 0; r < 8; ++r)
        xp[r] = out[(size_t)(mtile * 16 + r + hi * 8) * (S_ * H_)
                    + (size_t)t * H_ + n];

    v8f acc = {};
    if (t > 0) {
        const float*  hrow = out + (size_t)(mtile * 16 + lm) * (S_ * H_)
                                 + (size_t)(t - 1) * H_;
        const __bf16* bp   = Whh_bf + (size_t)n * H_ + koffB;

        v16bf a = make_a_frag(hrow, 0, koffA);
        v16bf b = *(const v16bf*)(bp);
        for (int k0 = 32; k0 < H_; k0 += 32) {
            v16bf an = make_a_frag(hrow, k0, koffA);
            v16bf bn = *(const v16bf*)(bp + k0);
            acc = __builtin_amdgcn_wmma_f32_16x16x32_bf16(false, a, false, b,
                                                          (short)0, acc, false, false);
            a = an; b = bn;
        }
        acc = __builtin_amdgcn_wmma_f32_16x16x32_bf16(false, a, false, b,
                                                      (short)0, acc, false, false);
    }

#pragma unroll
    for (int r = 0; r < 8; ++r) {
        size_t idx = (size_t)(mtile * 16 + r + hi * 8) * (S_ * H_)
                   + (size_t)t * H_ + n;
        out[idx] = fast_tanh(acc[r] + xp[r]);
    }
}

// ---------------------------------------------------------------------------
// Launch: prep -> big GEMM into d_out -> 1024 sequential step kernels.
// Workspace layout: [W_ih bf16 512KB][W_hh bf16 512KB][bias f32 2KB].
// ---------------------------------------------------------------------------
extern "C" void kernel_launch(void* const* d_in, const int* in_sizes, int n_in,
                              void* d_out, int out_size, void* d_ws, size_t ws_size,
                              hipStream_t stream) {
    const float* x   = (const float*)d_in[0];
    const float* Wih = (const float*)d_in[1];
    const float* Whh = (const float*)d_in[2];
    const float* bih = (const float*)d_in[3];
    const float* bhh = (const float*)d_in[4];
    float* out = (float*)d_out;

    __bf16* Wih_bf = (__bf16*)d_ws;
    __bf16* Whh_bf = Wih_bf + (size_t)H_ * I_;
    float*  bias   = (float*)(Whh_bf + (size_t)H_ * H_);

    elman_prep_kernel<<<(H_ * I_ + 255) / 256, 256, 0, stream>>>(
        Wih, Whh, bih, bhh, Wih_bf, Whh_bf, bias);

    elman_xproj_kernel<<<(B_ * S_) / 16, 256, 0, stream>>>(x, Wih_bf, bias, out);

    for (int t = 0; t < S_; ++t)
        elman_step_kernel<<<16, 256, 0, stream>>>(Whh_bf, out, t);
}
